// MovieLensNetwork_64888365908361
// MI455X (gfx1250) — matrix operand order repaired and checked by reading, not compiled, not run on hardware
//
#include <hip/hip_runtime.h>
#include <hip/hip_bf16.h>
#include <math.h>

// Problem constants (match reference)
#define R_    5
#define NU_   50000
#define NM_   20000
#define FIN_  256
#define HID_  128
#define E_    100000
#define D_    640     // R_*HID_

typedef __bf16 bf16;
typedef __attribute__((ext_vector_type(16))) __bf16 v16bf;
typedef __attribute__((ext_vector_type(8)))  float  v8f;

union AFrag { uint4 q[2]; v16bf v; };

__device__ __forceinline__ float bf16lo(unsigned v) {
  union { unsigned u; float f; } c; c.u = v << 16; return c.f;
}
__device__ __forceinline__ float bf16hi(unsigned v) {
  union { unsigned u; float f; } c; c.u = v & 0xffff0000u; return c.f;
}

// ---------------------------------------------------------------------------
// K0a: pack the 10 SAGE weight blocks [Ws;Wn] (512x128 each) into bf16,
// transposed to K-contiguous: Wt[p][col][k].  p = r*2 + (0:movie,1:user).
// ---------------------------------------------------------------------------
__global__ __launch_bounds__(256) void pack_sage_weights_kernel(
    const float* __restrict__ Wsu, const float* __restrict__ Wnu,
    const float* __restrict__ Wsm, const float* __restrict__ Wnm,
    bf16* __restrict__ Wt)
{
  int idx = blockIdx.x * 256 + threadIdx.x;        // < 10*128*512 = 655360
  int k   = idx & 511;
  int col = (idx >> 9) & 127;
  int p   = idx >> 16;                             // 128*512 = 65536
  int r   = p >> 1;
  const float* Wself  = (p & 1) ? Wsm : Wsu;
  const float* Wneigh = (p & 1) ? Wnm : Wnu;
  const float* W = (k < FIN_) ? Wself : Wneigh;
  Wt[idx] = (bf16)W[(size_t)r * FIN_ * HID_ + (size_t)(k & (FIN_ - 1)) * HID_ + col];
}

// ---------------------------------------------------------------------------
// K0b: decoders f32 -> bf16 (layout [k][d][j] already K(=j)-contiguous)
// ---------------------------------------------------------------------------
__global__ __launch_bounds__(256) void pack_dec_kernel(
    const float* __restrict__ dec, bf16* __restrict__ dec_bf)
{
  int idx = blockIdx.x * 256 + threadIdx.x;        // < 5*640*640 = 2048000
  dec_bf[idx] = (bf16)dec[idx];
}

// ---------------------------------------------------------------------------
// K1: segment mean numerator + degree via float atomics. One wave per edge.
// ---------------------------------------------------------------------------
__global__ __launch_bounds__(256) void scatter_mean_kernel(
    const float* __restrict__ srcfeat, const int* __restrict__ src_idx,
    const int* __restrict__ dst_idx, float* __restrict__ agg,
    float* __restrict__ deg, int nedges)
{
  int gtid = blockIdx.x * 256 + threadIdx.x;
  int e    = gtid >> 5;
  int lane = threadIdx.x & 31;
  if (e >= nedges) return;
  int s = src_idx[e];
  int d = dst_idx[e];
  const float* srow = srcfeat + (size_t)s * FIN_;
  float*       arow = agg     + (size_t)d * FIN_;
#pragma unroll
  for (int i = 0; i < FIN_ / 32; ++i)
    atomicAdd(arow + lane + 32 * i, srow[lane + 32 * i]);
  if (lane == 0) atomicAdd(deg + d, 1.0f);
}

// ---------------------------------------------------------------------------
// K2: H = relu([dstfeat | agg/deg] @ [Ws;Wn] + b), row-L2-normalize,
// write into stacked embedding at column offset colOff (stride D_).
// Workgroup: 16 rows x 128 cols, 8 waves, one 16x16 WMMA tile per wave.
// B fragments: two b128 loads from pre-packed bf16 K-contiguous weights.
// ---------------------------------------------------------------------------
#define APAD  520   // 512 + 8 bf16 pad: rows 16B aligned, dword stride 260 -> bank-skewed
#define HPAD  132

__global__ __launch_bounds__(256) void sage_wmma_kernel(
    const float* __restrict__ dstfeat,
    const float* __restrict__ agg, const float* __restrict__ deg,
    const bf16* __restrict__ Wt,          // [128][512] bf16, K-contiguous
    const float* __restrict__ bias,
    float* __restrict__ emb, int colOff)
{
  __shared__ __align__(16) bf16  Atile[16 * APAD];
  __shared__ float Htile[16 * HPAD];
  __shared__ float rnorm[16];

  const int tid  = threadIdx.x;
  const int row0 = blockIdx.x * 16;

  // Stage A = [dstfeat | agg/max(deg,1)] as bf16
  for (int idx = tid; idx < 16 * 512; idx += 256) {
    int m = idx >> 9;
    int j = idx & 511;
    int row = row0 + m;
    float v;
    if (j < FIN_) {
      v = dstfeat[(size_t)row * FIN_ + j];
    } else {
      float rd = 1.0f / fmaxf(deg[row], 1.0f);
      v = agg[(size_t)row * FIN_ + (j - FIN_)] * rd;
    }
    Atile[m * APAD + j] = (bf16)v;
  }
  __syncthreads();

  const int wave = tid >> 5;
  const int lane = tid & 31;
  const int half = lane >> 4;
  const int l16  = lane & 15;
  const int col  = wave * 16 + l16;     // 0..127

  const bf16* wbase = Wt + (size_t)col * 512 + half * 16;

  v8f c = {};
#pragma unroll 4
  for (int k0 = 0; k0 < 512; k0 += 32) {
    // A fragment (16x32): lane holds row, K = k0+half*16 .. +15 (contiguous)
    AFrag a;
    const bf16* arow = &Atile[l16 * APAD + k0 + half * 16];
    a.q[0] = *(const uint4*)(arow);
    a.q[1] = *(const uint4*)(arow + 8);
    // B fragment (32x16): lane holds col, K = k0+half*16 .. +15 (contiguous)
    AFrag b;
    const bf16* wrow = wbase + k0;
    b.q[0] = *(const uint4*)(wrow);
    b.q[1] = *(const uint4*)(wrow + 8);
    c = __builtin_amdgcn_wmma_f32_16x16x32_bf16(false, a.v, false, b.v,
                                                (short)0, c, false, false);
  }

  // Epilogue: bias + relu into LDS (C/D layout: VGPR v -> M = v + half*8)
  float bcol = bias[col];
#pragma unroll
  for (int v = 0; v < 8; ++v) {
    int rowm = v + half * 8;
    Htile[rowm * HPAD + col] = fmaxf(c[v] + bcol, 0.0f);
  }
  __syncthreads();

  if (tid < 16) {
    float s = 0.0f;
    for (int j = 0; j < HID_; ++j) {
      float x = Htile[tid * HPAD + j];
      s += x * x;
    }
    rnorm[tid] = 1.0f / fmaxf(sqrtf(s), 1e-12f);
  }
  __syncthreads();

  for (int idx = tid; idx < 16 * HID_; idx += 256) {
    int m = idx >> 7;
    int j = idx & 127;
    emb[(size_t)(row0 + m) * D_ + colOff + j] = Htile[m * HPAD + j] * rnorm[m];
  }
}

// ---------------------------------------------------------------------------
// K3: Tm[k][m][d] = sum_j movie_emb[m][j] * dec[k][d][j]  (= M @ W_k^T)
// Grid: (NM/16, D/128, R). Output stored as bf16 (decode is gather-bound).
// ---------------------------------------------------------------------------
#define APAD2 648   // 640 + 8 bf16 pad

__global__ __launch_bounds__(256) void movie_transform_kernel(
    const float* __restrict__ memb, const bf16* __restrict__ dec_bf,
    bf16* __restrict__ Tm)
{
  __shared__ __align__(16) bf16 Atile[16 * APAD2];
  const int tid = threadIdx.x;
  const int m0  = blockIdx.x * 16;
  const int d0  = blockIdx.y * 128;
  const int k   = blockIdx.z;

  for (int idx = tid; idx < 16 * D_; idx += 256) {
    int m = idx / D_;
    int j = idx - m * D_;
    Atile[m * APAD2 + j] = (bf16)memb[(size_t)(m0 + m) * D_ + j];
  }
  __syncthreads();

  const int wave = tid >> 5;
  const int lane = tid & 31;
  const int half = lane >> 4;
  const int l16  = lane & 15;
  const int col  = d0 + wave * 16 + l16;
  const bf16* Bbase = dec_bf + ((size_t)k * D_ + col) * D_ + half * 16;

  v8f c = {};
#pragma unroll 4
  for (int k0 = 0; k0 < D_; k0 += 32) {
    AFrag a;
    const bf16* arow = &Atile[l16 * APAD2 + k0 + half * 16];
    a.q[0] = *(const uint4*)(arow);
    a.q[1] = *(const uint4*)(arow + 8);
    AFrag b;
    const bf16* wrow = Bbase + k0;
    b.q[0] = *(const uint4*)(wrow);
    b.q[1] = *(const uint4*)(wrow + 8);
    c = __builtin_amdgcn_wmma_f32_16x16x32_bf16(false, a.v, false, b.v,
                                                (short)0, c, false, false);
  }

  bf16* orow = Tm + ((size_t)k * NM_ + m0) * D_ + col;
#pragma unroll
  for (int v = 0; v < 8; ++v)
    orow[(size_t)(v + half * 8) * D_] = (bf16)c[v];
}

// ---------------------------------------------------------------------------
// K4: per edge: dots[k] = <user_emb[u] (f32), Tm[k][m] (bf16)>, log_softmax.
// One wave per (r, e); lane handles d = (lane+32*i)*4 .. +3, i = 0..4.
// ---------------------------------------------------------------------------
__global__ __launch_bounds__(256) void decode_kernel(
    const float* __restrict__ user_emb, const bf16* __restrict__ Tm,
    const int* __restrict__ edges_u, const int* __restrict__ edges_v,
    float* __restrict__ out)
{
  int gw   = (blockIdx.x * 256 + threadIdx.x) >> 5;
  int lane = threadIdx.x & 31;
  int r = gw / E_;
  int e = gw - r * E_;
  if (r >= R_) return;

  int u = edges_u[(size_t)r * E_ + e];
  int m = edges_v[(size_t)r * E_ + e];

  const float4* urow = (const float4*)(user_emb + (size_t)u * D_);
  float4 uf[5];
#pragma unroll
  for (int i = 0; i < 5; ++i) uf[i] = urow[lane + 32 * i];

  float acc[R_];
#pragma unroll
  for (int k = 0; k < R_; ++k) {
    const uint2* trow = (const uint2*)(Tm + ((size_t)k * NM_ + m) * D_);
    float s = 0.0f;
#pragma unroll
    for (int i = 0; i < 5; ++i) {
      uint2 t = trow[lane + 32 * i];      // 4 bf16
      s += uf[i].x * bf16lo(t.x) + uf[i].y * bf16hi(t.x)
         + uf[i].z * bf16lo(t.y) + uf[i].w * bf16hi(t.y);
    }
    acc[k] = s;
  }
#pragma unroll
  for (int k = 0; k < R_; ++k)
#pragma unroll
    for (int off = 16; off > 0; off >>= 1)
      acc[k] += __shfl_xor(acc[k], off, 32);

  if (lane == 0) {
    float mx = acc[0];
#pragma unroll
    for (int k = 1; k < R_; ++k) mx = fmaxf(mx, acc[k]);
    float se = 0.0f;
#pragma unroll
    for (int k = 0; k < R_; ++k) se += __expf(acc[k] - mx);
    float lse = mx + __logf(se);
    float* o = out + ((size_t)r * E_ + e) * R_;
#pragma unroll
    for (int k = 0; k < R_; ++k) o[k] = acc[k] - lse;
  }
}

// ---------------------------------------------------------------------------
extern "C" void kernel_launch(void* const* d_in, const int* in_sizes, int n_in,
                              void* d_out, int out_size, void* d_ws, size_t ws_size,
                              hipStream_t stream)
{
  const float* ufeats    = (const float*)d_in[0];
  const float* mfeats    = (const float*)d_in[1];
  const float* W_neigh_u = (const float*)d_in[2];
  const float* W_self_u  = (const float*)d_in[3];
  const float* b_u       = (const float*)d_in[4];
  const float* W_neigh_m = (const float*)d_in[5];
  const float* W_self_m  = (const float*)d_in[6];
  const float* b_m       = (const float*)d_in[7];
  const float* decoders  = (const float*)d_in[8];
  const int*   edges_u   = (const int*)d_in[9];
  const int*   edges_v   = (const int*)d_in[10];
  float* out = (float*)d_out;

  char* ws = (char*)d_ws;
  size_t off = 0;
  float* user_emb  = (float*)(ws + off); off += (size_t)NU_ * D_ * 4;          // 128 MB
  float* movie_emb = (float*)(ws + off); off += (size_t)NM_ * D_ * 4;          // 51.2 MB
  bf16*  Tm        = (bf16*) (ws + off); off += (size_t)R_ * NM_ * D_ * 2;     // 128 MB
  float* agg       = (float*)(ws + off); off += (size_t)NU_ * FIN_ * 4;        // 51.2 MB
  float* deg       = (float*)(ws + off); off += ((size_t)NU_ * 4 + 255) & ~(size_t)255;
  bf16*  Wt        = (bf16*) (ws + off); off += (size_t)10 * HID_ * 512 * 2;   // 1.31 MB
  bf16*  dec_bf    = (bf16*) (ws + off); off += (size_t)R_ * D_ * D_ * 2;      // 4.1 MB
  (void)ws_size; (void)in_sizes; (void)n_in; (void)out_size;

  dim3 blk(256);
  const int scat_blocks = (E_ * 32) / 256;  // 12500

  pack_sage_weights_kernel<<<(10 * HID_ * 512) / 256, blk, 0, stream>>>(
      W_self_u, W_neigh_u, W_self_m, W_neigh_m, Wt);
  pack_dec_kernel<<<(R_ * D_ * D_) / 256, blk, 0, stream>>>(decoders, dec_bf);

  for (int r = 0; r < R_; ++r) {
    // ---- movie embeddings for relation r: users -> movies ----
    hipMemsetAsync(agg, 0, (size_t)NM_ * FIN_ * 4, stream);
    hipMemsetAsync(deg, 0, (size_t)NM_ * 4, stream);
    scatter_mean_kernel<<<scat_blocks, blk, 0, stream>>>(
        ufeats, edges_u + (size_t)r * E_, edges_v + (size_t)r * E_, agg, deg, E_);
    sage_wmma_kernel<<<NM_ / 16, blk, 0, stream>>>(
        mfeats, agg, deg, Wt + (size_t)(r * 2 + 0) * HID_ * 512,
        b_u + (size_t)r * HID_, movie_emb, r * HID_);

    // ---- user embeddings for relation r: movies -> users ----
    hipMemsetAsync(agg, 0, (size_t)NU_ * FIN_ * 4, stream);
    hipMemsetAsync(deg, 0, (size_t)NU_ * 4, stream);
    scatter_mean_kernel<<<scat_blocks, blk, 0, stream>>>(
        mfeats, edges_v + (size_t)r * E_, edges_u + (size_t)r * E_, agg, deg, E_);
    sage_wmma_kernel<<<NU_ / 16, blk, 0, stream>>>(
        ufeats, agg, deg, Wt + (size_t)(r * 2 + 1) * HID_ * 512,
        b_m + (size_t)r * HID_, user_emb, r * HID_);
  }

  movie_transform_kernel<<<dim3(NM_ / 16, D_ / 128, R_), blk, 0, stream>>>(
      movie_emb, dec_bf, Tm);

  decode_kernel<<<(R_ * E_ * 32) / 256, blk, 0, stream>>>(
      user_emb, Tm, edges_u, edges_v, out);
}
